// LSTMDecoder_35261681500452
// MI455X (gfx1250) — matrix operand order repaired and checked by reading, not compiled, not run on hardware
//
#include <hip/hip_runtime.h>
#include <math.h>

// ---------------- Problem constants ----------------
#define Bn   32
#define Tn   64
#define TS   63        // T-1 steps
#define Hn   512
#define Vn   32000
#define G4H  2048      // 4*H
#define MROWS 2016     // TS*Bn
#define MT    126      // MROWS/16
#define KK    16       // Hn/32 k-steps of 32
#define NTW  4         // N-tiles staged per block in k3 (64 cols, 64KB LDS)

// ---------------- WMMA types ----------------
typedef __attribute__((ext_vector_type(16))) __bf16          v16bf;
typedef __attribute__((ext_vector_type(8)))  float           v8f;
typedef __attribute__((ext_vector_type(8)))  unsigned short  ushort8;
typedef __attribute__((ext_vector_type(16))) unsigned short  ushort16;

union U16x16 { ushort16 u; v16bf b; };

__device__ __forceinline__ v8f wmma_bf16(v16bf a, v16bf b, v8f c) {
  // D = A(16x32 bf16) * B(32x16 bf16) + C(16x16 f32)
  return __builtin_amdgcn_wmma_f32_16x16x32_bf16(
      /*neg_a=*/false, a, /*neg_b=*/false, b,
      /*c_mod=*/(short)0, c, /*reuse_a=*/false, /*reuse_b=*/false);
}

// f32 -> bf16 round-to-nearest-even (bit pattern as u16)
__device__ __forceinline__ unsigned short f2bf(float f) {
  unsigned int u = __float_as_uint(f);
  u += 0x7FFFu + ((u >> 16) & 1u);
  return (unsigned short)(u >> 16);
}

// A fragment (16x32 bf16), base = &A[row0][k0], row-major, leading dim ld.
// Layout used here (contiguous k, not ISA table order): lane l holds row m=l&15; half=l>>4 selects K chunk
// {half*16..half*16+15} (elems 0..15), contiguous, matching load_b_frag.
__device__ __forceinline__ v16bf load_a_frag(const unsigned short* base, int ld, int lane) {
  int m = lane & 15, half = lane >> 4;
  const unsigned short* r = base + (size_t)m * ld;
  ushort8 c0 = *(const ushort8*)(r + half * 16);
  ushort8 c1 = *(const ushort8*)(r + half * 16 + 8);
  U16x16 f;
#pragma unroll
  for (int i = 0; i < 8; ++i) { f.u[i] = c0[i]; f.u[i + 8] = c1[i]; }
  return f.b;
}

// B fragment = W^T tile (32x16): B[k][n] = W[n0+n][k0+k].
// base = &W[n0][k0], ld = leading dim of W. lane l: n=l&15, khalf=l>>4,
// elems 0..15 are 16 contiguous K values at k0 + khalf*16.
__device__ __forceinline__ v16bf load_b_frag(const unsigned short* base, int ld, int lane) {
  int n = lane & 15, kh = lane >> 4;
  const unsigned short* r = base + (size_t)n * ld + kh * 16;
  ushort8 c0 = *(const ushort8*)(r);
  ushort8 c1 = *(const ushort8*)(r + 8);
  U16x16 f;
#pragma unroll
  for (int i = 0; i < 8; ++i) { f.u[i] = c0[i]; f.u[i + 8] = c1[i]; }
  return f.b;
}

__device__ __forceinline__ float sigm(float x) { return 1.0f / (1.0f + __expf(-x)); }

// ---------------- Workspace layout (bytes, 256-aligned) ----------------
#define OFF_WIH   0u                     // 2048*1024 bf16  (4,194,304 B)
#define OFF_WHH   4194304u               // 2048*512  bf16  (2,097,152 B)
#define OFF_WOUT  6291456u               // 32000*512 bf16  (32,768,000 B)
#define OFF_ENCB  39059456u              // 32*512    bf16
#define OFF_XEMB  39092224u              // 2016*512  bf16
#define OFF_GXB   41156608u              // 2048      f32 (b_ih+b_hh)
#define OFF_GX    41164800u              // 2016*2048 f32
#define OFF_HBF   57679872u              // 2016*512  bf16
// total ~59.7 MB

// ============ Kernel 0: weight conversion + embedding gather ============
__global__ void k0_prep(const float* __restrict__ enc, const int* __restrict__ cap,
                        const float* __restrict__ emb,
                        const float* __restrict__ Wih, const float* __restrict__ Whh,
                        const float* __restrict__ bih, const float* __restrict__ bhh,
                        const float* __restrict__ Wout,
                        unsigned short* Wih_b, unsigned short* Whh_b,
                        unsigned short* Wout_b, unsigned short* encb,
                        unsigned short* X, float* gxb) {
  size_t tid = (size_t)blockIdx.x * blockDim.x + threadIdx.x;
  size_t stride = (size_t)gridDim.x * blockDim.x;
  for (size_t i = tid; i < (size_t)G4H * 2 * Hn; i += stride) Wih_b[i] = f2bf(Wih[i]);
  for (size_t i = tid; i < (size_t)G4H * Hn;     i += stride) Whh_b[i] = f2bf(Whh[i]);
  for (size_t i = tid; i < (size_t)Vn * Hn;      i += stride) Wout_b[i] = f2bf(Wout[i]);
  for (size_t i = tid; i < (size_t)Bn * Hn;      i += stride) encb[i] = f2bf(enc[i]);
  for (size_t i = tid; i < (size_t)MROWS * Hn;   i += stride) {
    size_t r = i >> 9;            // row = t*32 + b
    int j = (int)(i & 511);
    int t = (int)(r >> 5), b = (int)(r & 31);
    int tok = cap[b * Tn + t];
    X[i] = f2bf(emb[(size_t)tok * Hn + j]);
  }
  for (size_t i = tid; i < G4H; i += stride) gxb[i] = bih[i] + bhh[i];
}

// ============ Kernel 1: gx = [enc|emb] @ W_ih^T + (b_ih+b_hh) ============
// grid (4, 126), block 256 (8 waves). Wave: 1 M-tile x 4 N-tiles.
__global__ void k1_gx(const unsigned short* __restrict__ Wih_b,
                      const unsigned short* __restrict__ X,
                      const unsigned short* __restrict__ encb,
                      const float* __restrict__ gxb, float* __restrict__ gx) {
  int lane = threadIdx.x & 31, wave = threadIdx.x >> 5;
  int mt = blockIdx.y;
  int nt0 = blockIdx.x * 32 + wave * 4;
  v8f z = {0, 0, 0, 0, 0, 0, 0, 0};
  v8f acc[4] = {z, z, z, z};

  const unsigned short* Arow  = X + (size_t)(mt * 16) * Hn;           // emb half
  const unsigned short* ArowE = encb + (size_t)((mt * 16) & 31) * Hn; // enc half (b-periodic)

  for (int kk = 0; kk < KK; ++kk) {                 // emb part: cols [512,1024) of W_ih
    v16bf a = load_a_frag(Arow + kk * 32, Hn, lane);
#pragma unroll
    for (int i = 0; i < 4; ++i) {
      v16bf b = load_b_frag(Wih_b + (size_t)((nt0 + i) * 16) * (2 * Hn) + Hn + kk * 32,
                            2 * Hn, lane);
      acc[i] = wmma_bf16(a, b, acc[i]);
    }
  }
  for (int kk = 0; kk < KK; ++kk) {                 // enc part: cols [0,512)
    v16bf a = load_a_frag(ArowE + kk * 32, Hn, lane);
#pragma unroll
    for (int i = 0; i < 4; ++i) {
      v16bf b = load_b_frag(Wih_b + (size_t)((nt0 + i) * 16) * (2 * Hn) + kk * 32,
                            2 * Hn, lane);
      acc[i] = wmma_bf16(a, b, acc[i]);
    }
  }
  int half = lane >> 4, nl = lane & 15;
#pragma unroll
  for (int i = 0; i < 4; ++i) {
    int col = (nt0 + i) * 16 + nl;
    float bias = gxb[col];
#pragma unroll
    for (int v = 0; v < 8; ++v) {
      int row = mt * 16 + half * 8 + v;
      gx[(size_t)row * G4H + col] = acc[i][v] + bias;
    }
  }
}

// ============ Kernel 2: sequential LSTM recurrence (1 WGP, 32 waves) ============
// h kept as bf16 in LDS (ds_load_b128 A-fragments); c kept in registers in the
// exact C-fragment layout. 64 tile-units (2 Mtiles x 32 j-tiles), 2 per wave;
// each unit accumulates all four gates (i,f,g,o) so the nonlinearity is in-lane.
__global__ void __launch_bounds__(1024)
k2_lstm(const unsigned short* __restrict__ Whh_b, const float* __restrict__ gx,
        unsigned short* __restrict__ hbf) {
  __shared__ unsigned short hsh[Bn * Hn];  // 32 KB: h in bf16
  int tid = threadIdx.x;
  int lane = tid & 31, wave = tid >> 5;
  int half = lane >> 4, nl = lane & 15;

  for (int i = tid; i < Bn * Hn; i += 1024) hsh[i] = 0;  // h0 = 0
  v8f z = {0, 0, 0, 0, 0, 0, 0, 0};
  v8f cst[2] = {z, z};                                    // c0 = 0
  __syncthreads();

  for (int t = 0; t < TS; ++t) {
    v8f acc[2][4];
    // ---- phase 1: gates_pre = h @ W_hh^T (reads hsh) ----
    for (int q = 0; q < 2; ++q) {
      int u = wave * 2 + q;
      int mt = u >> 5, j = u & 31;
#pragma unroll
      for (int g = 0; g < 4; ++g) acc[q][g] = z;
      for (int kk = 0; kk < KK; ++kk) {
        v16bf a = load_a_frag(&hsh[(mt * 16) * Hn + kk * 32], Hn, lane);  // LDS
#pragma unroll
        for (int g = 0; g < 4; ++g) {
          v16bf b = load_b_frag(Whh_b + (size_t)(g * Hn + j * 16) * Hn + kk * 32, Hn, lane);
          acc[q][g] = wmma_bf16(a, b, acc[q][g]);
        }
      }
    }
    __syncthreads();  // all reads of h(t-1) complete before overwriting
    // ---- phase 2: elementwise LSTM cell, write h(t) ----
    for (int q = 0; q < 2; ++q) {
      int u = wave * 2 + q;
      int mt = u >> 5, j = u & 31;
      int col = j * 16 + nl;
#pragma unroll
      for (int v = 0; v < 8; ++v) {
        int br = mt * 16 + half * 8 + v;  // batch row
        const float* gxr = gx + (size_t)(t * Bn + br) * G4H;
        float gi = acc[q][0][v] + gxr[0 * Hn + col];
        float gf = acc[q][1][v] + gxr[1 * Hn + col];
        float gg = acc[q][2][v] + gxr[2 * Hn + col];
        float go = acc[q][3][v] + gxr[3 * Hn + col];
        float c = sigm(gf) * cst[q][v] + sigm(gi) * tanhf(gg);
        float h = sigm(go) * tanhf(c);
        cst[q][v] = c;
        unsigned short hb = f2bf(h);
        hsh[br * Hn + col] = hb;
        hbf[(size_t)(t * Bn + br) * Hn + col] = hb;
      }
    }
    __syncthreads();  // h(t) visible before next step's reads
  }
}

// ============ Kernel 3: logits = H @ W_out^T + b_out ============
// grid (500, 16), block 256 (8 waves). Block stages 64 W_out rows (64 KB)
// into LDS ONCE via global_load_async_to_lds_b128 (ASYNCcnt), then its 8
// waves each compute a DIFFERENT M-tile against the staged B -> W_out L2
// traffic drops from 126x to 16x (4.1 GB -> 0.53 GB). Logits stream to HBM.
__global__ void __launch_bounds__(256)
k3_logits(const unsigned short* __restrict__ Wout_b,
          const unsigned short* __restrict__ Hbf,
          const float* __restrict__ bout, float* __restrict__ out) {
  __shared__ unsigned short Bsh[NTW * 16 * Hn];  // 64 KB
  int lane = threadIdx.x & 31, wave = threadIdx.x >> 5;
  int n0row = blockIdx.x * (NTW * 16);           // first W_out row of this group
  int mt = blockIdx.y * 8 + wave;                // wave-private M-tile (may be >= MT)

  // ---- cooperative async stage: W_out rows [n0row, n0row+64) -> LDS ----
  {
    unsigned lds_base = (unsigned)(uintptr_t)(&Bsh[0]);  // low 32 bits = LDS offset
    unsigned long long gbase =
        (unsigned long long)(uintptr_t)(Wout_b + (size_t)n0row * Hn);
#pragma unroll
    for (int it = 0; it < 16; ++it) {
      unsigned boff = (unsigned)(it * 256 + threadIdx.x) * 16u;  // byte offset
      asm volatile("global_load_async_to_lds_b128 %0, %1, %2"
                   :: "v"(lds_base + boff), "v"(boff), "s"(gbase)
                   : "memory");
    }
    asm volatile("s_wait_asynccnt 0x0" ::: "memory");  // our async writes done
  }
  __syncthreads();  // all waves' staged data visible

  if (mt < MT) {  // wave-uniform: EXEC all-1s inside (WMMA requirement)
    v8f z = {0, 0, 0, 0, 0, 0, 0, 0};
    v8f acc[NTW] = {z, z, z, z};
    const unsigned short* Arow = Hbf + (size_t)(mt * 16) * Hn;

    for (int kk = 0; kk < KK; ++kk) {
      if (kk < KK - 1)
        __builtin_prefetch(Arow + (kk + 1) * 32, 0, 1);
      v16bf a = load_a_frag(Arow + kk * 32, Hn, lane);
#pragma unroll
      for (int i = 0; i < NTW; ++i) {
        v16bf b = load_b_frag(&Bsh[(i * 16) * Hn + kk * 32], Hn, lane);  // LDS
        acc[i] = wmma_bf16(a, b, acc[i]);
      }
    }
    int half = lane >> 4, nl = lane & 15;
#pragma unroll
    for (int i = 0; i < NTW; ++i) {
      int col = n0row + i * 16 + nl;
      float bias = bout[col];
#pragma unroll
      for (int v = 0; v < 8; ++v) {
        int row = mt * 16 + half * 8 + v;      // row = t*32 + b
        int t = row >> 5, b = row & 31;
        out[(size_t)(b * TS + t) * Vn + col] = acc[i][v] + bias;  // (B, T-1, V)
      }
    }
  }
}

// ============ Kernel 4: per-row max + argmax over V ============
__global__ void k4_argmax(const float* __restrict__ logits,
                          float* __restrict__ probs, float* __restrict__ preds) {
  __shared__ float smax[256];
  __shared__ int sidx[256];
  int r = blockIdx.x;
  const float* row = logits + (size_t)r * Vn;
  float m = -INFINITY; int mi = 0;
  for (int v = threadIdx.x; v < Vn; v += 256) {
    float x = row[v];
    if (x > m) { m = x; mi = v; }
  }
  smax[threadIdx.x] = m; sidx[threadIdx.x] = mi;
  __syncthreads();
  for (int s = 128; s > 0; s >>= 1) {
    if (threadIdx.x < s) {
      float o = smax[threadIdx.x + s]; int oi = sidx[threadIdx.x + s];
      if (o > smax[threadIdx.x] ||
          (o == smax[threadIdx.x] && oi < sidx[threadIdx.x])) {
        smax[threadIdx.x] = o; sidx[threadIdx.x] = oi;
      }
    }
    __syncthreads();
  }
  if (threadIdx.x == 0) { probs[r] = smax[0]; preds[r] = (float)sidx[0]; }
}

// ============ Launch ============
extern "C" void kernel_launch(void* const* d_in, const int* in_sizes, int n_in,
                              void* d_out, int out_size, void* d_ws, size_t ws_size,
                              hipStream_t stream) {
  const float* enc  = (const float*)d_in[0];
  const int*   cap  = (const int*)d_in[1];
  const float* emb  = (const float*)d_in[2];
  const float* Wih  = (const float*)d_in[3];
  const float* Whh  = (const float*)d_in[4];
  const float* bih  = (const float*)d_in[5];
  const float* bhh  = (const float*)d_in[6];
  const float* Wout = (const float*)d_in[7];
  const float* bout = (const float*)d_in[8];
  float* out = (float*)d_out;

  char* ws = (char*)d_ws;
  unsigned short* Wih_b  = (unsigned short*)(ws + OFF_WIH);
  unsigned short* Whh_b  = (unsigned short*)(ws + OFF_WHH);
  unsigned short* Wout_b = (unsigned short*)(ws + OFF_WOUT);
  unsigned short* encb   = (unsigned short*)(ws + OFF_ENCB);
  unsigned short* X      = (unsigned short*)(ws + OFF_XEMB);
  float*          gxb    = (float*)(ws + OFF_GXB);
  float*          gxbuf  = (float*)(ws + OFF_GX);
  unsigned short* Hbf    = (unsigned short*)(ws + OFF_HBF);

  k0_prep<<<1024, 256, 0, stream>>>(enc, cap, emb, Wih, Whh, bih, bhh, Wout,
                                    Wih_b, Whh_b, Wout_b, encb, X, gxb);
  k1_gx<<<dim3(4, 126), 256, 0, stream>>>(Wih_b, X, encb, gxb, gxbuf);
  k2_lstm<<<1, 1024, 0, stream>>>(Whh_b, gxbuf, Hbf);
  k3_logits<<<dim3(500, 16), 256, 0, stream>>>(Wout_b, Hbf, bout, out);

  float* probs = out + (size_t)Bn * TS * Vn;         // 64,512,000
  float* preds = probs + (size_t)Bn * TS;            // +2016
  k4_argmax<<<MROWS, 256, 0, stream>>>(out, probs, preds);
}
